// MultiScaleQuantizer_41068477284658
// MI455X (gfx1250) — matrix operand order repaired and checked by reading, not compiled, not run on hardware
//
#include <hip/hip_runtime.h>
#include <hip/hip_bf16.h>

typedef __attribute__((ext_vector_type(16))) _Float16 v16h;
typedef __attribute__((ext_vector_type(8)))  float    v8f;
typedef __attribute__((ext_vector_type(4)))  float    v4f;

#define HW    16
#define CDIM  32
#define BATCH 256
#define NCODE 4096
#define NPIX  (BATCH * CDIM * HW * HW)   // 2,097,152

// ---------------------------------------------------------------------------
// init: f_rest = z, f_hat (in d_out) = 0, loss = 0
// ---------------------------------------------------------------------------
__global__ void k_init(const float* __restrict__ z, float* __restrict__ f_rest,
                       float* __restrict__ f_hat, float* __restrict__ loss) {
  int i = blockIdx.x * blockDim.x + threadIdx.x;
  if (i < NPIX) { f_rest[i] = z[i]; f_hat[i] = 0.f; }
  if (i == 0) *loss = 0.f;
}

// ---------------------------------------------------------------------------
// prep: enorm[v] = ||embed[v]||^2 ; repack codebook to f16 in WMMA B layout.
// B is 32(K) x 16(N) per code tile t: lane n (0..15) holds col n, K=0..15 in
// v16h elems 0..15; lane n+16 holds col n, K=16..31.
// ---------------------------------------------------------------------------
__global__ void k_prep_embed(const float* __restrict__ embed,
                             float* __restrict__ enorm,
                             _Float16* __restrict__ Bp) {
  int v = blockIdx.x * blockDim.x + threadIdx.x;
  if (v >= NCODE) return;
  const float* e = embed + v * CDIM;
  int t = v >> 4, col = v & 15;
  float s = 0.f;
  for (int k = 0; k < CDIM; ++k) {
    float x = e[k];
    s += x * x;
    int lane = (k < 16) ? col : (col + 16);
    Bp[t * 512 + lane * 16 + (k & 15)] = (_Float16)x;
  }
  enorm[v] = s;
}

// ---------------------------------------------------------------------------
// bicubic row-stochastic matrices W_pn (16 x pn) for pn = 1,2,4,8
// (torch upsample_bicubic2d, align_corners=False, A=-0.75)
// ---------------------------------------------------------------------------
__device__ __forceinline__ float cubicw(float s) {
  const float Ac = -0.75f;
  s = fabsf(s);
  if (s <= 1.f) return ((Ac + 2.f) * s - (Ac + 3.f)) * s * s + 1.f;
  if (s <  2.f) return (((s - 5.f) * s + 8.f) * s - 4.f) * Ac;
  return 0.f;
}

__global__ void k_weights(float* __restrict__ wbuf) {
  int tid = threadIdx.x;          // 64 threads: (scale s, out-row i)
  if (tid >= 64) return;
  int s = tid >> 4, i = tid & 15;
  int pn = 1 << s;
  int off = (s == 0) ? 0 : (s == 1) ? 16 : (s == 2) ? 48 : 112;
  float* W = wbuf + off + i * pn;
  for (int j = 0; j < pn; ++j) W[j] = 0.f;
  float scale = (float)pn / 16.f;
  float x  = (i + 0.5f) * scale - 0.5f;
  float x0 = floorf(x);
  float t  = x - x0;
  int ix0 = (int)x0;
  for (int o = -1; o <= 2; ++o) {
    int idx = ix0 + o;
    idx = idx < 0 ? 0 : (idx > pn - 1 ? pn - 1 : idx);
    W[idx] += cubicw(t - (float)o);
  }
}

// ---------------------------------------------------------------------------
// area downsample f_rest(b,c,16,16) -> Zsc row-major (N=b*pn*pn, 32)
// pn=16 (kh=1) degenerates to the layout transpose, matching z_scale=f_rest.
// ---------------------------------------------------------------------------
__global__ void k_down(const float* __restrict__ f_rest, float* __restrict__ Zsc,
                       int pn) {
  int tid = blockIdx.x * blockDim.x + threadIdx.x;
  int ntot = BATCH * pn * pn * CDIM;
  if (tid >= ntot) return;
  int c = tid & 31;
  int n = tid >> 5;
  int x = n % pn;
  int y = (n / pn) % pn;
  int b = n / (pn * pn);
  int kh = HW / pn;
  const float* base = f_rest + ((b * CDIM + c) * HW + y * kh) * HW + x * kh;
  float s = 0.f;
  for (int dy = 0; dy < kh; ++dy)
    for (int dx = 0; dx < kh; ++dx)
      s += base[dy * HW + dx];
  Zsc[tid] = s / (float)(kh * kh);
}

// ---------------------------------------------------------------------------
// VQ core: one wave per 16-row tile. D = Z(16x32) * E^T(32x16) via
// v_wmma_f32_16x16x32_f16, running argmin of ||e||^2 - 2 z.e over 256 code
// tiles, shfl-xor min-reduce per 16-lane half, gather winners in f32,
// per-tile squared-error partial (deterministic).
// ---------------------------------------------------------------------------
__global__ void __launch_bounds__(256)
k_vq(const float* __restrict__ Zsc, const _Float16* __restrict__ Bp,
     const float* __restrict__ enorm, const float* __restrict__ embed,
     float* __restrict__ Zq, float* __restrict__ idx_out,
     float* __restrict__ losspart, int ntiles) {
  int wave = blockIdx.x * 8 + (threadIdx.x >> 5);
  int lane = threadIdx.x & 31;
  if (wave >= ntiles) return;          // uniform per wave: EXEC stays all-ones
  int rowt = wave * 16;
  int col  = lane & 15;
  int kb   = (lane >> 4) * 16;         // A layout: half-wave K offset

  // A: 16x32 f16. lane<16: K = kb..kb+15 (kb=0);
  //               lane>=16: same with kb=16.
  const float* zp = Zsc + (rowt + col) * CDIM + kb;
  v4f z0 = *(const v4f*)(zp);
  v4f z1 = *(const v4f*)(zp + 4);
  v4f z2 = *(const v4f*)(zp + 8);
  v4f z3 = *(const v4f*)(zp + 12);
  v16h a;
#pragma unroll
  for (int j = 0; j < 4; ++j) {
    a[j]      = (_Float16)z0[j];
    a[4 + j]  = (_Float16)z1[j];
    a[8 + j]  = (_Float16)z2[j];
    a[12 + j] = (_Float16)z3[j];
  }

  float bestd[8];
  int   besti[8];
#pragma unroll
  for (int r = 0; r < 8; ++r) { bestd[r] = 3.4e38f; besti[r] = 0; }

  for (int t = 0; t < 256; t += 2) {   // 2-way unroll: hide WMMA->VALU latency
    v16h b0 = *(const v16h*)(Bp + (t    ) * 512 + lane * 16);
    v16h b1 = *(const v16h*)(Bp + (t + 1) * 512 + lane * 16);
    v8f c0 = {}, c1 = {};
    c0 = __builtin_amdgcn_wmma_f32_16x16x32_f16(false, a, false, b0, (short)0, c0, false, false);
    c1 = __builtin_amdgcn_wmma_f32_16x16x32_f16(false, a, false, b1, (short)0, c1, false, false);
    float en0 = enorm[t * 16 + col];
    float en1 = enorm[(t + 1) * 16 + col];
    int code0 = t * 16 + col;
    int code1 = code0 + 16;
#pragma unroll
    for (int r = 0; r < 8; ++r) {
      float d0 = __builtin_fmaf(-2.f, c0[r], en0);
      if (d0 < bestd[r]) { bestd[r] = d0; besti[r] = code0; }
      float d1 = __builtin_fmaf(-2.f, c1[r], en1);
      if (d1 < bestd[r]) { bestd[r] = d1; besti[r] = code1; }
    }
  }

  // min-reduce across N (16 lanes of each half); masks 1..8 never cross halves
#pragma unroll
  for (int mask = 8; mask >= 1; mask >>= 1) {
#pragma unroll
    for (int r = 0; r < 8; ++r) {
      float od = __shfl_xor(bestd[r], mask);
      int   oi = __shfl_xor(besti[r], mask);
      if (od < bestd[r] || (od == bestd[r] && oi < besti[r])) {
        bestd[r] = od; besti[r] = oi;
      }
    }
  }
  // lanes 0..15 now hold winners for rows r (r=0..7); lanes 16..31 rows 8+r.

  if ((lane & 15) == 0) {
    int base = rowt + (lane >> 4) * 8;
#pragma unroll
    for (int r = 0; r < 8; ++r) idx_out[base + r] = (float)besti[r];
  }

  // broadcast the 16 winners; m is a compile-time index after unroll
  int widx[16];
#pragma unroll
  for (int r = 0; r < 8; ++r) {
    widx[r]     = __shfl(besti[r], 0);
    widx[8 + r] = __shfl(besti[r], 16);
  }

  // cooperative f32 gather of winning codes + squared-error partial
  float sq = 0.f;
#pragma unroll
  for (int m = 0; m < 16; ++m) {
    float v = embed[widx[m] * CDIM + lane];
    int gi  = (rowt + m) * CDIM + lane;
    Zq[gi]  = v;
    float d = v - Zsc[gi];
    sq = __builtin_fmaf(d, d, sq);
  }
#pragma unroll
  for (int mask = 16; mask >= 1; mask >>= 1) sq += __shfl_xor(sq, mask);
  if (lane == 0) losspart[wave] = sq;
}

// ---------------------------------------------------------------------------
// deterministic loss finalize: fixed strided accumulation + fixed tree
// loss += scalefac * sum(losspart[0..ntiles))
// ---------------------------------------------------------------------------
__global__ void k_loss(const float* __restrict__ losspart, int ntiles,
                       float scalefac, float* __restrict__ loss) {
  __shared__ float sm[256];
  int tid = threadIdx.x;
  float s = 0.f;
  for (int i = tid; i < ntiles; i += 256) s += losspart[i];
  sm[tid] = s;
  __syncthreads();
  for (int st = 128; st > 0; st >>= 1) {
    if (tid < st) sm[tid] += sm[tid + st];
    __syncthreads();
  }
  if (tid == 0) *loss += sm[0] * scalefac;
}

// ---------------------------------------------------------------------------
// bicubic upsample (pn -> 16) + f_hat += ; f_rest -=
// ---------------------------------------------------------------------------
__global__ void k_up(const float* __restrict__ Zq, const float* __restrict__ W,
                     float* __restrict__ f_hat, float* __restrict__ f_rest,
                     int pn) {
  int tid = blockIdx.x * blockDim.x + threadIdx.x;
  if (tid >= NPIX) return;
  int P = tid & 15;
  int O = (tid >> 4) & 15;
  int c = (tid >> 8) & 31;
  int b = tid >> 13;
  float val = 0.f;
  for (int h = 0; h < pn; ++h) {
    float wh = W[O * pn + h];
    const float* zrow = Zq + ((b * pn + h) * pn) * CDIM + c;
    for (int w = 0; w < pn; ++w)
      val = __builtin_fmaf(wh * W[P * pn + w], zrow[w * CDIM], val);
  }
  f_hat[tid] += val;
  f_rest[tid] -= val;
}

// last scale: z_q = zq (no resample); only f_hat needed afterwards
__global__ void k_add_final(const float* __restrict__ Zq, float* __restrict__ f_hat) {
  int tid = blockIdx.x * blockDim.x + threadIdx.x;
  if (tid >= NPIX) return;
  int P = tid & 15;
  int O = (tid >> 4) & 15;
  int c = (tid >> 8) & 31;
  int b = tid >> 13;
  int n = (b * HW + O) * HW + P;
  f_hat[tid] += Zq[n * CDIM + c];
}

// ---------------------------------------------------------------------------
// launch
// ---------------------------------------------------------------------------
extern "C" void kernel_launch(void* const* d_in, const int* in_sizes, int n_in,
                              void* d_out, int out_size, void* d_ws, size_t ws_size,
                              hipStream_t stream) {
  const float* z     = (const float*)d_in[0];
  const float* embed = (const float*)d_in[1];
  float* out = (float*)d_out;

  char* ws = (char*)d_ws;
  float*    f_rest   = (float*)(ws);                    //  8 MB
  float*    Zsc      = (float*)(ws + 8388608);          //  8 MB
  float*    Zq       = (float*)(ws + 16777216);         //  8 MB
  _Float16* Bp       = (_Float16*)(ws + 25165824);      // 256 KB (B-layout f16 codebook)
  float*    enorm    = (float*)(ws + 25427968);         // 16 KB
  float*    wbuf     = (float*)(ws + 25444352);         // bicubic matrices
  float*    losspart = (float*)(ws + 25445376);         // up to 4096 partials

  float* f_hat    = out;                  // (256,32,16,16)
  float* idx_base = out + 2097152;        // 87,296 index floats
  float* loss     = out + 2184448;        // scalar

  k_init<<<(NPIX + 255) / 256, 256, 0, stream>>>(z, f_rest, f_hat, loss);
  k_prep_embed<<<NCODE / 256, 256, 0, stream>>>(embed, enorm, Bp);
  k_weights<<<1, 64, 0, stream>>>(wbuf);

  const int pns[5]  = {1, 2, 4, 8, 16};
  const int woff[4] = {0, 16, 48, 112};
  long idxoff = 0;
  for (int i = 0; i < 5; ++i) {
    int pn = pns[i];
    int N = BATCH * pn * pn;           // rows at this scale
    int ntiles = N / 16;
    k_down<<<(N * CDIM + 255) / 256, 256, 0, stream>>>(f_rest, Zsc, pn);
    k_vq<<<(ntiles + 7) / 8, 256, 0, stream>>>(Zsc, Bp, enorm, embed, Zq,
                                               idx_base + idxoff, losspart, ntiles);
    // loss_i = (1+beta)*MSE = 1.25 * sum / (N*32); final loss averaged over 5
    k_loss<<<1, 256, 0, stream>>>(losspart, ntiles,
                                  1.25f / ((float)N * 32.f) / 5.f, loss);
    if (i < 4)
      k_up<<<(NPIX + 255) / 256, 256, 0, stream>>>(Zq, wbuf + woff[i], f_hat, f_rest, pn);
    else
      k_add_final<<<(NPIX + 255) / 256, 256, 0, stream>>>(Zq, f_hat);
    idxoff += N;
  }
}
